// TransformerBlock_21285857919384
// MI455X (gfx1250) — matrix operand order repaired and checked by reading, not compiled, not run on hardware
//
#include <hip/hip_runtime.h>
#include <hip/hip_bf16.h>
#include <math.h>

// ---------------------------------------------------------------------------
// Transformer block for MI455X (gfx1250): bf16 WMMA everywhere, fp32 accum.
// D=1024, H=16, dk=64, B=2, T=2048, DFF=4096.
// GEMM uses double-buffered LDS tiles fed by CDNA5 async global->LDS loads
// (ASYNCcnt) when the toolchain exposes them.
// ---------------------------------------------------------------------------

typedef __attribute__((ext_vector_type(16))) __bf16        v16bf;
typedef __attribute__((ext_vector_type(8)))  float         v8f;
typedef __attribute__((ext_vector_type(4)))  unsigned int  u32x4;
typedef __attribute__((ext_vector_type(4)))  int           i32x4;
typedef __attribute__((ext_vector_type(4)))  float         f32x4;

#define D_MODEL 1024
#define NHEADS  16
#define DKH     64
#define BATCH   2
#define SEQ     2048
#define SROWS   (BATCH*SEQ)
#define DFF     4096

enum { FLAG_BIAS = 1, FLAG_GELU = 2, FLAG_RES = 4, FLAG_BF16 = 8, FLAG_VT = 16 };

#if defined(__has_builtin)
#  if __has_builtin(__builtin_amdgcn_global_load_async_to_lds_b128) && \
      __has_builtin(__builtin_amdgcn_s_wait_asynccnt)
#    define USE_ASYNC_LDS 1
#  endif
#endif
#ifndef USE_ASYNC_LDS
#  define USE_ASYNC_LDS 0
#endif

// Builtin expects int4 pointers: (int4 AS1* src, int4 AS3* dst, imm off, imm cpol)
#define GLOBAL_V4(p) ((__attribute__((address_space(1))) i32x4*)(p))
#define LDS_V4(p)    ((__attribute__((address_space(3)))  i32x4*)(p))

__device__ __forceinline__ unsigned short f2bf(float f) {
  unsigned int u = __float_as_uint(f);
  u += 0x7fffu + ((u >> 16) & 1u);          // round-to-nearest-even
  return (unsigned short)(u >> 16);
}

union FragAB { v16bf v; u32x4 q[2]; unsigned short h[16]; };
union FragC  { v8f   v; float f[8]; };

// ---------------------------------------------------------------------------
// Weight convert+transpose: W[K][N] fp32 -> Wt[N][K] bf16 (so WMMA B-frags
// become contiguous 128-bit loads along K).
// ---------------------------------------------------------------------------
__global__ __launch_bounds__(256)
void transpose_w_bf16(const float* __restrict__ W, unsigned short* __restrict__ Wt,
                      int K, int N) {
  __shared__ float tile[32][33];
  const int tx = threadIdx.x, ty = threadIdx.y;
  const int k0 = blockIdx.y * 32, n0 = blockIdx.x * 32;
#pragma unroll
  for (int i = 0; i < 4; ++i)
    tile[ty + 8 * i][tx] = W[(size_t)(k0 + ty + 8 * i) * N + n0 + tx];
  __syncthreads();
#pragma unroll
  for (int i = 0; i < 4; ++i)
    Wt[(size_t)(n0 + ty + 8 * i) * K + k0 + tx] = f2bf(tile[tx][ty + 8 * i]);
}

// ---------------------------------------------------------------------------
// Row layernorm: fp32 in -> bf16 out. One block (256 thr) per row of 1024.
// ---------------------------------------------------------------------------
__global__ __launch_bounds__(256)
void layernorm_bf16(const float* __restrict__ X, const float* __restrict__ g,
                    const float* __restrict__ bt, unsigned short* __restrict__ Y) {
  __shared__ float rs[256], rq[256];
  const int row = blockIdx.x, tid = threadIdx.x;
  const float* x = X + (size_t)row * D_MODEL + tid * 4;
  f32x4 v = *(const f32x4*)x;
  rs[tid] = v[0] + v[1] + v[2] + v[3];
  rq[tid] = v[0]*v[0] + v[1]*v[1] + v[2]*v[2] + v[3]*v[3];
  __syncthreads();
  for (int o = 128; o > 0; o >>= 1) {
    if (tid < o) { rs[tid] += rs[tid + o]; rq[tid] += rq[tid + o]; }
    __syncthreads();
  }
  const float mean = rs[0] * (1.f / D_MODEL);
  const float var  = rq[0] * (1.f / D_MODEL) - mean * mean;
  const float rstd = rsqrtf(var + 1e-5f);
#pragma unroll
  for (int i = 0; i < 4; ++i) {
    const int c = tid * 4 + i;
    Y[(size_t)row * D_MODEL + c] = f2bf((v[i] - mean) * rstd * g[c] + bt[c]);
  }
}

// ---------------------------------------------------------------------------
// bf16 WMMA GEMM: C[M,N] = A[M,K] (bf16, row-major) x Bt[N,K] (bf16).
// Block = 128 thr (4 waves), tile 64x64, K-step 32, double-buffered LDS with
// 40-half row pitch. Tile k+1 is fetched (async global->LDS when available)
// while tile k is multiplied. Fused epilogue: bias / exact GELU / fp32
// residual / fp32|bf16|transposed-V output.
// ---------------------------------------------------------------------------
__global__ __launch_bounds__(128)
void gemm_bf16_wmma(const unsigned short* __restrict__ A, int lda,
                    const unsigned short* __restrict__ Bt, int ldb,
                    float* __restrict__ outF, unsigned short* __restrict__ outH, int ldo,
                    const float* __restrict__ bias,
                    const float* __restrict__ resid, int ldr,
                    int K, int flags) {
  __shared__ unsigned short As[2][64 * 40];
  __shared__ unsigned short Bs[2][64 * 40];
  const int tid  = threadIdx.x;
  const int lane = tid & 31;
  const int wave = tid >> 5;
  const int wm = wave >> 1, wn = wave & 1;
  const int m0 = blockIdx.y * 64;
  const int n0 = blockIdx.x * 64;
  const int hi = (lane & 16) ? 1 : 0;
  const int koffA = hi ? 8 : 0;     // A-frag K offset per lane half
  const int boffB = hi ? 8 : 0;     // B-frag K offset per lane half
  const int lrow = lane & 15;

  // per-thread cooperative-load coordinates (2 x 16B chunks of each tile)
  const int li0 = tid * 2,     r0 = li0 >> 2, c0 = (li0 & 3) * 8;
  const int li1 = tid * 2 + 1, r1 = li1 >> 2, c1 = (li1 & 3) * 8;

  FragC acc[2][2];
#pragma unroll
  for (int a = 0; a < 2; ++a)
#pragma unroll
    for (int b = 0; b < 2; ++b)
#pragma unroll
      for (int j = 0; j < 8; ++j) acc[a][b].f[j] = 0.f;

  auto issue_tile = [&](int k0, int buf) {
#if USE_ASYNC_LDS
    __builtin_amdgcn_global_load_async_to_lds_b128(
        GLOBAL_V4(A  + (size_t)(m0 + r0) * lda + k0 + c0), LDS_V4(&As[buf][r0 * 40 + c0]), 0, 0);
    __builtin_amdgcn_global_load_async_to_lds_b128(
        GLOBAL_V4(A  + (size_t)(m0 + r1) * lda + k0 + c1), LDS_V4(&As[buf][r1 * 40 + c1]), 0, 0);
    __builtin_amdgcn_global_load_async_to_lds_b128(
        GLOBAL_V4(Bt + (size_t)(n0 + r0) * ldb + k0 + c0), LDS_V4(&Bs[buf][r0 * 40 + c0]), 0, 0);
    __builtin_amdgcn_global_load_async_to_lds_b128(
        GLOBAL_V4(Bt + (size_t)(n0 + r1) * ldb + k0 + c1), LDS_V4(&Bs[buf][r1 * 40 + c1]), 0, 0);
#else
    *(u32x4*)&As[buf][r0 * 40 + c0] = *(const u32x4*)&A [(size_t)(m0 + r0) * lda + k0 + c0];
    *(u32x4*)&As[buf][r1 * 40 + c1] = *(const u32x4*)&A [(size_t)(m0 + r1) * lda + k0 + c1];
    *(u32x4*)&Bs[buf][r0 * 40 + c0] = *(const u32x4*)&Bt[(size_t)(n0 + r0) * ldb + k0 + c0];
    *(u32x4*)&Bs[buf][r1 * 40 + c1] = *(const u32x4*)&Bt[(size_t)(n0 + r1) * ldb + k0 + c1];
#endif
  };

  issue_tile(0, 0);
  const int nk = K >> 5;
  for (int kt = 0; kt < nk; ++kt) {
    const int buf = kt & 1;
#if USE_ASYNC_LDS
    __builtin_amdgcn_s_wait_asynccnt(0);     // own tile-kt async writes done
#endif
    __syncthreads();                          // all waves' tile kt visible; prev reads done
    if (kt + 1 < nk) issue_tile((kt + 1) * 32, buf ^ 1);  // overlaps compute below

    FragAB af[2], bfg[2];
#pragma unroll
    for (int fm = 0; fm < 2; ++fm) {
      const int r = wm * 32 + fm * 16 + lrow;
      af[fm].q[0] = *(const u32x4*)&As[buf][r * 40 + koffA];
      af[fm].q[1] = *(const u32x4*)&As[buf][r * 40 + koffA + 16];
    }
#pragma unroll
    for (int fn = 0; fn < 2; ++fn) {
      const int r = wn * 32 + fn * 16 + lrow;
      bfg[fn].q[0] = *(const u32x4*)&Bs[buf][r * 40 + boffB];
      bfg[fn].q[1] = *(const u32x4*)&Bs[buf][r * 40 + boffB + 16];
    }
#pragma unroll
    for (int fm = 0; fm < 2; ++fm)
#pragma unroll
      for (int fn = 0; fn < 2; ++fn)
        acc[fm][fn].v = __builtin_amdgcn_wmma_f32_16x16x32_bf16(
            false, af[fm].v, false, bfg[fn].v, (short)0, acc[fm][fn].v, false, false);
  }

#pragma unroll
  for (int fm = 0; fm < 2; ++fm) {
#pragma unroll
    for (int fn = 0; fn < 2; ++fn) {
      const int rbase = m0 + wm * 32 + fm * 16 + (hi ? 8 : 0);
      const int col   = n0 + wn * 32 + fn * 16 + lrow;
      const float bi = (flags & FLAG_BIAS) ? bias[col] : 0.f;
#pragma unroll
      for (int j = 0; j < 8; ++j) {
        const int row = rbase + j;
        float v = acc[fm][fn].f[j] + bi;
        if (flags & FLAG_GELU) v = 0.5f * v * (1.f + erff(v * 0.70710678118f));
        if (flags & FLAG_RES)  v += resid[(size_t)row * ldr + col];
        if (flags & FLAG_BF16) {
          outH[(size_t)row * ldo + col] = f2bf(v);
        } else if (flags & FLAG_VT) {   // V: write transposed per batch: [b][d][t]
          const int bb = row >> 11, tt = row & (SEQ - 1);
          outH[(((size_t)bb * D_MODEL) + col) * SEQ + tt] = f2bf(v);
        } else {
          outF[(size_t)row * ldo + col] = v;
        }
      }
    }
  }
}

// ---------------------------------------------------------------------------
// Flash attention, causal. Block = 4 waves; each wave owns 16 query rows of
// one (b,h), loops over 32-key tiles. Q/K bf16 row-major [S,D]; V transposed
// bf16 [b][d][t] so P*V B-frags are contiguous. Online softmax via shfl.
// ---------------------------------------------------------------------------
__global__ __launch_bounds__(128)
void attn_wmma(const unsigned short* __restrict__ Qm,
               const unsigned short* __restrict__ Km,
               const unsigned short* __restrict__ Vt,
               unsigned short* __restrict__ Om) {
  __shared__ unsigned short Pl[4 * 16 * 40];
  const int tid  = threadIdx.x;
  const int lane = tid & 31;
  const int wave = tid >> 5;
  const int hi   = (lane & 16) ? 1 : 0;
  const int lrow = lane & 15;
  const int b = blockIdx.y / NHEADS;
  const int h = blockIdx.y % NHEADS;
  const int qbase = blockIdx.x * 64 + wave * 16;
  const int hbase = h * DKH;
  const size_t bT = (size_t)b * SEQ;

  FragAB qa[2];                               // Q A-frags, d-chunks of 32
  {
    const unsigned short* qp = Qm + (bT + qbase + lrow) * D_MODEL + hbase;
#pragma unroll
    for (int dc = 0; dc < 2; ++dc) {
      const unsigned short* p = qp + dc * 32 + (hi ? 8 : 0);
      qa[dc].q[0] = *(const u32x4*)p;
      qa[dc].q[1] = *(const u32x4*)(p + 16);
    }
  }

  float m[8], l[8];
  FragC accO[4];
#pragma unroll
  for (int j = 0; j < 8; ++j) { m[j] = -1e30f; l[j] = 0.f; }
#pragma unroll
  for (int dt = 0; dt < 4; ++dt)
#pragma unroll
    for (int j = 0; j < 8; ++j) accO[dt].f[j] = 0.f;

  unsigned short* Pw = &Pl[wave * 16 * 40];
  const int ntiles = (qbase + 16 + 31) >> 5;   // causal: keys 0..qbase+15

  for (int tk = 0; tk < ntiles; ++tk) {
    const int kb = tk * 32;
    FragC s0, s1;
#pragma unroll
    for (int j = 0; j < 8; ++j) { s0.f[j] = 0.f; s1.f[j] = 0.f; }
#pragma unroll
    for (int dc = 0; dc < 2; ++dc) {
      FragAB kf0, kf1;
      const unsigned short* kp = Km + (bT + kb + lrow) * D_MODEL + hbase + dc * 32 + (hi ? 8 : 0);
      kf0.q[0] = *(const u32x4*)kp;      kf0.q[1] = *(const u32x4*)(kp + 16);
      const unsigned short* kp1 = kp + (size_t)16 * D_MODEL;
      kf1.q[0] = *(const u32x4*)kp1;     kf1.q[1] = *(const u32x4*)(kp1 + 16);
      s0.v = __builtin_amdgcn_wmma_f32_16x16x32_bf16(false, qa[dc].v, false, kf0.v, (short)0, s0.v, false, false);
      s1.v = __builtin_amdgcn_wmma_f32_16x16x32_bf16(false, qa[dc].v, false, kf1.v, (short)0, s1.v, false, false);
    }
    // online softmax update (rows spread over j and lane halves; cols over lanes)
#pragma unroll
    for (int j = 0; j < 8; ++j) {
      const int qrow = qbase + j + (hi ? 8 : 0);
      const int kc0 = kb + lrow, kc1 = kb + 16 + lrow;
      float v0 = (kc0 <= qrow) ? s0.f[j] * 0.125f : -1e30f;   // 1/sqrt(64)
      float v1 = (kc1 <= qrow) ? s1.f[j] * 0.125f : -1e30f;
      float mx = fmaxf(v0, v1);
#pragma unroll
      for (int o = 1; o < 16; o <<= 1) mx = fmaxf(mx, __shfl_xor(mx, o, 16));
      const float mnew = fmaxf(m[j], mx);
      const float corr = __expf(m[j] - mnew);
      m[j] = mnew;
      const float p0 = __expf(v0 - mnew);
      const float p1 = __expf(v1 - mnew);
      float rs = p0 + p1;
#pragma unroll
      for (int o = 1; o < 16; o <<= 1) rs += __shfl_xor(rs, o, 16);
      l[j] = l[j] * corr + rs;
#pragma unroll
      for (int dt = 0; dt < 4; ++dt) accO[dt].f[j] *= corr;
      const int pr = (j + (hi ? 8 : 0)) * 40 + lrow;
      Pw[pr]      = f2bf(p0);
      Pw[pr + 16] = f2bf(p1);
    }
    // P A-frag from LDS (wave-private region; DS ops in-order within wave)
    FragAB pa;
    {
      const unsigned short* p = Pw + lrow * 40 + (hi ? 8 : 0);
      pa.q[0] = *(const u32x4*)p;
      pa.q[1] = *(const u32x4*)(p + 16);
    }
    // O += P * V   (V transposed: row = d, contiguous t)
#pragma unroll
    for (int dt = 0; dt < 4; ++dt) {
      FragAB vf;
      const unsigned short* vp =
          Vt + ((size_t)b * D_MODEL + hbase + dt * 16 + lrow) * SEQ + kb + (hi ? 8 : 0);
      vf.q[0] = *(const u32x4*)vp;  vf.q[1] = *(const u32x4*)(vp + 16);
      accO[dt].v = __builtin_amdgcn_wmma_f32_16x16x32_bf16(
          false, pa.v, false, vf.v, (short)0, accO[dt].v, false, false);
    }
  }
  // finalize: O /= l, write bf16 [S,D]
#pragma unroll
  for (int dt = 0; dt < 4; ++dt) {
#pragma unroll
    for (int j = 0; j < 8; ++j) {
      const int row = qbase + j + (hi ? 8 : 0);
      const int col = hbase + dt * 16 + lrow;
      Om[(bT + row) * D_MODEL + col] = f2bf(accO[dt].f[j] / l[j]);
    }
  }
}

// ---------------------------------------------------------------------------
// Host orchestration
// ---------------------------------------------------------------------------
extern "C" void kernel_launch(void* const* d_in, const int* in_sizes, int n_in,
                              void* d_out, int out_size, void* d_ws, size_t ws_size,
                              hipStream_t stream) {
  (void)in_sizes; (void)n_in; (void)out_size; (void)ws_size;
  const float* H    = (const float*)d_in[0];
  const float* Wq   = (const float*)d_in[1];
  const float* Wk   = (const float*)d_in[2];
  const float* Wv   = (const float*)d_in[3];
  const float* Wo   = (const float*)d_in[4];
  const float* ln1g = (const float*)d_in[5];
  const float* ln1b = (const float*)d_in[6];
  const float* ln2g = (const float*)d_in[7];
  const float* ln2b = (const float*)d_in[8];
  const float* W1   = (const float*)d_in[9];
  const float* b1   = (const float*)d_in[10];
  const float* W2   = (const float*)d_in[11];
  const float* b2   = (const float*)d_in[12];

  char* ws = (char*)d_ws;
  size_t off = 0;
  auto take = [&](size_t bytes) -> char* {
    char* p = ws + off;
    off += (bytes + 255) & ~(size_t)255;
    return p;
  };
  unsigned short* WqT = (unsigned short*)take((size_t)D_MODEL * D_MODEL * 2);
  unsigned short* WkT = (unsigned short*)take((size_t)D_MODEL * D_MODEL * 2);
  unsigned short* WvT = (unsigned short*)take((size_t)D_MODEL * D_MODEL * 2);
  unsigned short* WoT = (unsigned short*)take((size_t)D_MODEL * D_MODEL * 2);
  unsigned short* W1T = (unsigned short*)take((size_t)D_MODEL * DFF * 2);
  unsigned short* W2T = (unsigned short*)take((size_t)D_MODEL * DFF * 2);
  unsigned short* Xln = (unsigned short*)take((size_t)SROWS * D_MODEL * 2); // reused: attn out
  unsigned short* Qb  = (unsigned short*)take((size_t)SROWS * D_MODEL * 2); // reused: X2
  unsigned short* Kb  = (unsigned short*)take((size_t)SROWS * D_MODEL * 2);
  unsigned short* Vt  = (unsigned short*)take((size_t)BATCH * D_MODEL * SEQ * 2);
  float*          H1  = (float*)         take((size_t)SROWS * D_MODEL * 4);
  unsigned short* Hff = (unsigned short*)take((size_t)SROWS * DFF * 2);

  const dim3 tb(32, 8);
  // Weights -> bf16, transposed [N,K]
  transpose_w_bf16<<<dim3(D_MODEL/32, D_MODEL/32), tb, 0, stream>>>(Wq, WqT, D_MODEL, D_MODEL);
  transpose_w_bf16<<<dim3(D_MODEL/32, D_MODEL/32), tb, 0, stream>>>(Wk, WkT, D_MODEL, D_MODEL);
  transpose_w_bf16<<<dim3(D_MODEL/32, D_MODEL/32), tb, 0, stream>>>(Wv, WvT, D_MODEL, D_MODEL);
  transpose_w_bf16<<<dim3(D_MODEL/32, D_MODEL/32), tb, 0, stream>>>(Wo, WoT, D_MODEL, D_MODEL);
  transpose_w_bf16<<<dim3(DFF/32,     D_MODEL/32), tb, 0, stream>>>(W1, W1T, D_MODEL, DFF);
  transpose_w_bf16<<<dim3(D_MODEL/32, DFF/32),     tb, 0, stream>>>(W2, W2T, DFF, D_MODEL);

  // LN1
  layernorm_bf16<<<SROWS, 256, 0, stream>>>(H, ln1g, ln1b, Xln);

  // Q, K (bf16 row-major), V (bf16 transposed [b][d][t])
  gemm_bf16_wmma<<<dim3(D_MODEL/64, SROWS/64), 128, 0, stream>>>(
      Xln, D_MODEL, WqT, D_MODEL, nullptr, Qb, D_MODEL, nullptr, nullptr, 0, D_MODEL, FLAG_BF16);
  gemm_bf16_wmma<<<dim3(D_MODEL/64, SROWS/64), 128, 0, stream>>>(
      Xln, D_MODEL, WkT, D_MODEL, nullptr, Kb, D_MODEL, nullptr, nullptr, 0, D_MODEL, FLAG_BF16);
  gemm_bf16_wmma<<<dim3(D_MODEL/64, SROWS/64), 128, 0, stream>>>(
      Xln, D_MODEL, WvT, D_MODEL, nullptr, Vt, 0,       nullptr, nullptr, 0, D_MODEL, FLAG_VT);

  // Causal flash attention -> Xln (bf16, heads merged)
  attn_wmma<<<dim3(SEQ/64, BATCH*NHEADS), 128, 0, stream>>>(Qb, Kb, Vt, Xln);

  // O projection + residual H -> H1 (fp32)
  gemm_bf16_wmma<<<dim3(D_MODEL/64, SROWS/64), 128, 0, stream>>>(
      Xln, D_MODEL, WoT, D_MODEL, H1, nullptr, D_MODEL, nullptr, H, D_MODEL, D_MODEL, FLAG_RES);

  // LN2 -> X2 (reuse Qb)
  layernorm_bf16<<<SROWS, 256, 0, stream>>>(H1, ln2g, ln2b, Qb);

  // MLP up + GELU -> Hff (bf16)
  gemm_bf16_wmma<<<dim3(DFF/64, SROWS/64), 128, 0, stream>>>(
      Qb, D_MODEL, W1T, D_MODEL, nullptr, Hff, DFF, b1, nullptr, 0, D_MODEL,
      FLAG_BIAS | FLAG_GELU | FLAG_BF16);

  // MLP down + bias + residual H1 -> d_out (fp32)
  gemm_bf16_wmma<<<dim3(D_MODEL/64, SROWS/64), 128, 0, stream>>>(
      Hff, DFF, W2T, DFF, (float*)d_out, nullptr, D_MODEL, b2, H1, D_MODEL, DFF,
      FLAG_BIAS | FLAG_RES);
}
